// AmygdalaIndexer_90855738180037
// MI455X (gfx1250) — matrix operand order repaired and checked by reading, not compiled, not run on hardware
//
#include <hip/hip_runtime.h>

// CDNA5 / gfx1250, wave32. bf16 WMMA (v_wmma_f32_16x16x32_bf16), f32 accumulate.
typedef __attribute__((ext_vector_type(16))) __bf16 v16bf;
typedef __attribute__((ext_vector_type(8)))  __bf16 v8bf;
typedef __attribute__((ext_vector_type(8)))  float  v8f;

#define WMMA_BF16(a, b, c) \
  __builtin_amdgcn_wmma_f32_16x16x32_bf16(false, (a), false, (b), (short)0, (c), false, false)

constexpr int BATCH  = 4;
constexpr int SEQ    = 4096;
constexpr int NROWS  = BATCH * SEQ;   // 16384
constexpr int DMODEL = 1024;
constexpr int DL     = 64;            // latent dim
constexpr int QKW    = 2 * DL;        // q|k fused row width (128)
// softmax logits scale folded into log2 domain: (1/sqrt(64)) / 0.1 * log2(e)
constexpr float COEF = 1.25f * 1.4426950408889634f;

__device__ __forceinline__ v8f v8f_zero() {
  v8f z;
#pragma unroll
  for (int i = 0; i < 8; ++i) z[i] = 0.f;
  return z;
}

__device__ __forceinline__ v16bf cat16(v8bf lo, v8bf hi) {
  return __builtin_shufflevector(lo, hi, 0, 1, 2, 3, 4, 5, 6, 7,
                                 8, 9, 10, 11, 12, 13, 14, 15);
}

// A operand (16xK row m = lane&15, k-contiguous bf16): e -> k = 16*(e>>3)+8*half+(e&7)
__device__ __forceinline__ v16bf load_a_bf16(const __bf16* row, int half) {
  v8bf lo = *(const v8bf*)(row + 8 * half);
  v8bf hi = *(const v8bf*)(row + 16 + 8 * half);
  return cat16(lo, hi);
}
// B operand (Kx16 col n = lane&15, k-contiguous bf16): e -> k = 16*(e>>3)+8*half+(e&7)
__device__ __forceinline__ v16bf load_b_bf16(const __bf16* row, int half) {
  v8bf lo = *(const v8bf*)(row + 8 * half);
  v8bf hi = *(const v8bf*)(row + 16 + 8 * half);
  return cat16(lo, hi);
}
// A operand from f32 row (b128 loads + v_cvt_pk_bf16_f32)
__device__ __forceinline__ v16bf load_a_f32(const float* row, int half) {
  float4 q0 = *(const float4*)(row + 8 * half);
  float4 q1 = *(const float4*)(row + 8 * half + 4);
  float4 q2 = *(const float4*)(row + 16 + 8 * half);
  float4 q3 = *(const float4*)(row + 16 + 8 * half + 4);
  v16bf a;
  a[0] = (__bf16)q0.x;  a[1] = (__bf16)q0.y;  a[2]  = (__bf16)q0.z;  a[3]  = (__bf16)q0.w;
  a[4] = (__bf16)q1.x;  a[5] = (__bf16)q1.y;  a[6]  = (__bf16)q1.z;  a[7]  = (__bf16)q1.w;
  a[8] = (__bf16)q2.x;  a[9] = (__bf16)q2.y;  a[10] = (__bf16)q2.z;  a[11] = (__bf16)q2.w;
  a[12] = (__bf16)q3.x; a[13] = (__bf16)q3.y; a[14] = (__bf16)q3.z;  a[15] = (__bf16)q3.w;
  return a;
}
__device__ __forceinline__ v8bf pack8(const v8f& v) {
  v8bf r;
#pragma unroll
  for (int i = 0; i < 8; ++i) r[i] = (__bf16)v[i];
  return r;
}

// ---------------------------------------------------------------------------
// Prep: transpose + convert weights to bf16, k-contiguous per output column.
// ---------------------------------------------------------------------------
__global__ __launch_bounds__(256) void prep_wd_kernel(const float* __restrict__ Wd,
                                                      __bf16* __restrict__ WdT) {
  const int idx = blockIdx.x * 256 + threadIdx.x;  // 192*1024
  const int n = idx >> 10, k = idx & 1023;
  WdT[idx] = (__bf16)Wd[(size_t)k * 192 + n];
}
__global__ __launch_bounds__(256) void prep_wu_kernel(const float* __restrict__ Wu,
                                                      __bf16* __restrict__ WuT) {
  const int idx = blockIdx.x * 256 + threadIdx.x;  // 1024*64
  const int n = idx >> 6, k = idx & 63;
  WuT[idx] = (__bf16)Wu[(size_t)k * DMODEL + n];
}

// ---------------------------------------------------------------------------
// Kernel 1: qkv projection. One wave per 16-row M tile, all 12 N tiles.
// Writes q|k as bf16 rows (k-contig) and v transposed: vT[dl][NROWS] bf16.
// ---------------------------------------------------------------------------
__global__ __launch_bounds__(128) void qkv_proj_kernel(
    const float* __restrict__ x, const __bf16* __restrict__ WdT,
    const float* __restrict__ bd, __bf16* __restrict__ qk,
    __bf16* __restrict__ vT) {
  const int lane = threadIdx.x & 31;
  const int wave = threadIdx.x >> 5;
  const int mt   = blockIdx.x * 4 + wave;     // 1024 M tiles
  const int half = lane >> 4;
  const int lo   = lane & 15;
  const int m0   = mt * 16;

  const float* arow = x + (size_t)(m0 + lo) * DMODEL;

  v8f acc[12];
#pragma unroll
  for (int n = 0; n < 12; ++n) acc[n] = v8f_zero();

  for (int k0 = 0; k0 < DMODEL; k0 += 32) {
    __builtin_prefetch(arow + k0 + 256, 0, 1);
    const v16bf a = load_a_f32(arow + k0, half);
#pragma unroll
    for (int n = 0; n < 12; ++n) {
      const v16bf bm = load_b_bf16(WdT + (size_t)(n * 16 + lo) * DMODEL + k0, half);
      acc[n] = WMMA_BF16(a, bm, acc[n]);
    }
  }
  // q | k -> bf16 rows of 128 (scatter b16; row stride 128)
#pragma unroll
  for (int n = 0; n < 8; ++n) {
    const float bias = bd[n * 16 + lo];
#pragma unroll
    for (int r = 0; r < 8; ++r)
      qk[(size_t)(m0 + r + 8 * half) * QKW + n * 16 + lo] =
          (__bf16)(acc[n][r] + bias);
  }
  // v -> transposed vT[dl][m], 8 consecutive m per lane -> one b128 store
#pragma unroll
  for (int n = 8; n < 12; ++n) {
    const float bias = bd[n * 16 + lo];
    v8f t = acc[n];
#pragma unroll
    for (int r = 0; r < 8; ++r) t[r] += bias;
    const int dl = (n - 8) * 16 + lo;
    *(v8bf*)(vT + (size_t)dl * NROWS + m0 + 8 * half) = pack8(t);
  }
}

// ---------------------------------------------------------------------------
// Kernel 2: flash attention, one wave per 16-query tile, 32 keys per step.
// S^T = K·Q^T (softmax stats become per-lane scalars), ctx^T += V^T·P^T.
// All WMMA operands except P^T are direct b128 bf16 vector loads.
// ---------------------------------------------------------------------------
__global__ __launch_bounds__(128) void attention_kernel(
    const __bf16* __restrict__ qk, const __bf16* __restrict__ vT,
    const int* __restrict__ amask, __bf16* __restrict__ ctx) {
  const int lane = threadIdx.x & 31;
  const int wave = threadIdx.x >> 5;
  const int t    = blockIdx.x * 4 + wave;     // 1024 tiles = B * (S/16)
  const int b    = t >> 8;
  const int q0   = (t & 255) * 16;
  const int half = lane >> 4;
  const int lo   = lane & 15;
  const size_t rowbase = (size_t)b * SEQ;

  // Q as two B-matrices (latent chunks 0..31, 32..63)
  const __bf16* qrow = qk + (rowbase + q0 + lo) * QKW;
  const v16bf bq0 = load_b_bf16(qrow, half);
  const v16bf bq1 = load_b_bf16(qrow + 32, half);

  v8f o[4];
#pragma unroll
  for (int n = 0; n < 4; ++n) o[n] = v8f_zero();
  float M = -1e30f, L = 0.f;

  const int* am = amask + (size_t)b * SEQ;

  for (int kb = 0; kb < SEQ; kb += 32) {
    const unsigned mb = __builtin_amdgcn_ballot_w32(am[kb + lane] != 0);

    v8f s[2];
#pragma unroll
    for (int tl = 0; tl < 2; ++tl) {
      const __bf16* krow = qk + (rowbase + kb + tl * 16 + lo) * QKW + DL;
      v8f z = v8f_zero();
      z = WMMA_BF16(load_a_bf16(krow, half), bq0, z);
      s[tl] = WMMA_BF16(load_a_bf16(krow + 32, half), bq1, z);
    }

    float p0[8], p1[8];
    float vmax = -1e30f;
#pragma unroll
    for (int r = 0; r < 8; ++r) {
      const int bi = r + 8 * half;
      const float s0v = s[0][r] * COEF + (((mb >> bi) & 1u) ? 0.f : -1e30f);
      const float s1v = s[1][r] * COEF + (((mb >> (16 + bi)) & 1u) ? 0.f : -1e30f);
      p0[r] = s0v; p1[r] = s1v;
      vmax = fmaxf(vmax, fmaxf(s0v, s1v));
    }
    vmax = fmaxf(vmax, __shfl_xor(vmax, 16, 32));

    const float Mn    = fmaxf(M, vmax);
    const float alpha = exp2f(M - Mn);
    M = Mn;

    float rs = 0.f;
#pragma unroll
    for (int r = 0; r < 8; ++r) {
      const int bi = r + 8 * half;
      p0[r] = exp2f(p0[r] - M) * (float)((mb >> bi) & 1u);
      p1[r] = exp2f(p1[r] - M) * (float)((mb >> (16 + bi)) & 1u);
      rs += p0[r] + p1[r];
    }
    rs += __shfl_xor(rs, 16, 32);
    L = L * alpha + rs;
#pragma unroll
    for (int n = 0; n < 4; ++n)
#pragma unroll
      for (int r = 0; r < 8; ++r) o[n][r] *= alpha;

    // Build P^T (32 keys x 16 q) via packed-bf16 lane shuffles.
    v16bf bp;
#pragma unroll
    for (int i = 0; i < 8; ++i) {
      const int e = 2 * i;
      union Pk { unsigned u; __bf16 h[2]; } u0, u1, g;
      u0.h[0] = (__bf16)p0[e & 7]; u0.h[1] = (__bf16)p0[(e + 1) & 7];
      u1.h[0] = (__bf16)p1[e & 7]; u1.h[1] = (__bf16)p1[(e + 1) & 7];
      const int src = lo + (half ? 16 : 0);
      const int g0 = __shfl((int)u0.u, src, 32);
      const int g1 = __shfl((int)u1.u, src, 32);
      g.u = (unsigned)((e & 8) ? g1 : g0);
      bp[e] = g.h[0]; bp[e + 1] = g.h[1];
    }

    // ctx^T += V^T * P^T   (vT is dl-major -> key-contiguous A loads)
#pragma unroll
    for (int n = 0; n < 4; ++n) {
      const __bf16* vrow = vT + (size_t)(n * 16 + lo) * NROWS + rowbase + kb;
      o[n] = WMMA_BF16(load_a_bf16(vrow, half), bp, o[n]);
    }
  }

  const float inv = 1.f / L;  // per-lane (per q column)
#pragma unroll
  for (int n = 0; n < 4; ++n) {
    v8f tv = o[n];
#pragma unroll
    for (int r = 0; r < 8; ++r) tv[r] *= inv;
    // lane owns q col lo; latent n*16 + 8*half + (0..7) contiguous -> b128 store
    *(v8bf*)(ctx + (rowbase + q0 + lo) * DL + n * 16 + 8 * half) = pack8(tv);
  }
}

// ---------------------------------------------------------------------------
// Kernel 3: out = ctx @ Wu + bu.  One wave per (16-row M tile x 128-col group).
// ---------------------------------------------------------------------------
__global__ __launch_bounds__(128) void out_proj_kernel(
    const __bf16* __restrict__ ctx, const __bf16* __restrict__ WuT,
    const float* __restrict__ bu, float* __restrict__ out) {
  const int lane = threadIdx.x & 31;
  const int wave = threadIdx.x >> 5;
  const int t    = blockIdx.x * 4 + wave;     // 8192 = 1024 M tiles * 8 groups
  const int mt   = t >> 3;
  const int ng   = t & 7;
  const int m0   = mt * 16;
  const int n0   = ng * 128;
  const int half = lane >> 4;
  const int lo   = lane & 15;

  const __bf16* arow = ctx + (size_t)(m0 + lo) * DL;
  const v16bf a0 = load_a_bf16(arow, half);
  const v16bf a1 = load_a_bf16(arow + 32, half);

#pragma unroll
  for (int n = 0; n < 8; ++n) {
    const int ncol = n0 + n * 16 + lo;
    const __bf16* brow = WuT + (size_t)ncol * DL;
    v8f c = v8f_zero();
    c = WMMA_BF16(a0, load_b_bf16(brow, half), c);
    c = WMMA_BF16(a1, load_b_bf16(brow + 32, half), c);
    const float bias = bu[ncol];
#pragma unroll
    for (int r = 0; r < 8; ++r)
      out[(size_t)(m0 + r + 8 * half) * DMODEL + ncol] = c[r] + bias;
  }
}

// ---------------------------------------------------------------------------
extern "C" void kernel_launch(void* const* d_in, const int* in_sizes, int n_in,
                              void* d_out, int out_size, void* d_ws, size_t ws_size,
                              hipStream_t stream) {
  const float* x   = (const float*)d_in[0];
  const int*   am  = (const int*)d_in[1];
  const float* Wd  = (const float*)d_in[2];
  const float* bd  = (const float*)d_in[3];
  const float* Wu  = (const float*)d_in[4];
  const float* bu  = (const float*)d_in[5];
  float*       out = (float*)d_out;

  // bf16 workspace layout (all 16B aligned):
  //   qk  : NROWS*128         = 4 MB
  //   vT  : 64*NROWS          = 2 MB
  //   ctx : NROWS*64          = 2 MB
  //   WdT : 192*1024          = 384 KB
  //   WuT : 1024*64           = 128 KB
  __bf16* qk  = (__bf16*)d_ws;
  __bf16* vT  = qk  + (size_t)NROWS * QKW;
  __bf16* ctx = vT  + (size_t)DL * NROWS;
  __bf16* WdT = ctx + (size_t)NROWS * DL;
  __bf16* WuT = WdT + (size_t)192 * DMODEL;

  prep_wd_kernel<<<768, 256, 0, stream>>>(Wd, WdT);
  prep_wu_kernel<<<256, 256, 0, stream>>>(Wu, WuT);
  qkv_proj_kernel<<<256, 128, 0, stream>>>(x, WdT, bd, qk, vT);
  attention_kernel<<<256, 128, 0, stream>>>(qk, vT, am, ctx);
  out_proj_kernel<<<2048, 128, 0, stream>>>(ctx, WuT, bu, out);
}
